// PILayer_82386062672472
// MI455X (gfx1250) — matrix operand order repaired and checked by reading, not compiled, not run on hardware
//
#include <hip/hip_runtime.h>

typedef __attribute__((ext_vector_type(16))) _Float16 v16h;
typedef __attribute__((ext_vector_type(8)))  _Float16 v8h;
typedef __attribute__((ext_vector_type(8)))  float    v8f;
typedef __attribute__((ext_vector_type(4)))  float    v4f;

#define N_EDGES   640000
#define PROP_DIM  128
#define OUT_CH    64

// LDS layout (units: _Float16). Padded strides to avoid bank conflicts.
#define W0_STRIDE 264    // 256 K + 8 pad
#define W1_STRIDE 136    // 128 K + 8 pad
#define WO_STRIDE 136
#define H_STRIDE  136

#define W0_OFF 0
#define W1_OFF (W0_OFF + 128 * W0_STRIDE)          // 33792
#define WO_OFF (W1_OFF + 128 * W1_STRIDE)          // 51200
#define H_OFF  (WO_OFF + 256 * WO_STRIDE)          // 86016
#define H_PER_WAVE (32 * H_STRIDE)                 // 4352 halves (M=32, reused H0->H1)
#define SMEM_HALFS (H_OFF + 8 * H_PER_WAVE)        // 120832 halves = 241664 B

static __device__ __forceinline__ v16h cat8(v8h lo, v8h hi) {
  return __builtin_shufflevector(lo, hi, 0,1,2,3,4,5,6,7,8,9,10,11,12,13,14,15);
}

// Build one 16x32 f16 A fragment from a gathered f32 row (K run pair per lane).
static __device__ __forceinline__ v16h mk_afrag(const float* src, int k0) {
  v4f x0 = *(const v4f*)(src + k0);
  v4f x1 = *(const v4f*)(src + k0 + 4);
  v4f x2 = *(const v4f*)(src + k0 + 16);
  v4f x3 = *(const v4f*)(src + k0 + 20);
  v16h a;
  #pragma unroll
  for (int q = 0; q < 4; ++q) {
    a[q]      = (_Float16)x0[q];
    a[4 + q]  = (_Float16)x1[q];
    a[8 + q]  = (_Float16)x2[q];
    a[12 + q] = (_Float16)x3[q];
  }
  return a;
}

__global__ __launch_bounds__(256, 1)
void pilayer_kernel(const float* __restrict__ prop,
                    const int*   __restrict__ idx_i,
                    const int*   __restrict__ idx_j,
                    const float* __restrict__ basis,
                    const float* __restrict__ W0,
                    const float* __restrict__ b0,
                    const float* __restrict__ W1,
                    const float* __restrict__ b1,
                    const float* __restrict__ Wout,
                    float*       __restrict__ out)
{
  __shared__ _Float16 smem[SMEM_HALFS];
  const int tid = threadIdx.x;

  // ---- Stage weights once per WG: f32 -> f16, transposed to [N][K] ----
  for (int i = tid; i < 256 * 128; i += 256) {           // W0 [256K x 128N]
    int n = i & 127, k = i >> 7;
    smem[W0_OFF + n * W0_STRIDE + k] = (_Float16)W0[k * 128 + n];
  }
  for (int i = tid; i < 128 * 128; i += 256) {           // W1 [128K x 128N]
    int n = i & 127, k = i >> 7;
    smem[W1_OFF + n * W1_STRIDE + k] = (_Float16)W1[k * 128 + n];
  }
  for (int i = tid; i < 128 * 256; i += 256) {           // Wout [128K x 256N]
    int n = i & 255, k = i >> 8;
    smem[WO_OFF + n * WO_STRIDE + k] = (_Float16)Wout[k * 256 + n];
  }
  __syncthreads();

  const int lane = tid & 31;
  const int wv   = tid >> 5;          // wave in workgroup
  const int r    = lane & 15;         // A tile row index == B/C column index
  const int half = lane >> 4;         // lane-group select
  _Float16* __restrict__ hbuf = &smem[H_OFF + wv * H_PER_WAVE];

  // Per-lane bias preload (bias depends only on output column N)
  float bias0[8], bias1[8];
  #pragma unroll
  for (int nt = 0; nt < 8; ++nt) {
    bias0[nt] = b0[16 * nt + r];
    bias1[nt] = b1[16 * nt + r];
  }

  const int waveId = blockIdx.x * 8 + wv;
  const int nWaves = gridDim.x * 8;
  const int nTiles = N_EDGES / 32;    // M=32 edges per wave-tile

  for (int t = waveId; t < nTiles; t += nWaves) {
    const int base = t * 32;
    const int e0 = base + r;          // rows 0..15
    const int e1 = base + 16 + r;     // rows 16..31
    const float* rowi0 = prop + idx_i[e0] * PROP_DIM;
    const float* rowj0 = prop + idx_j[e0] * PROP_DIM;
    const float* rowi1 = prop + idx_i[e1] * PROP_DIM;
    const float* rowj1 = prop + idx_j[e1] * PROP_DIM;

    // ---- A fragments for both row-tiles, K = 0..255 (concat gather) ----
    v16h A0[8], A1[8];
    #pragma unroll
    for (int s = 0; s < 8; ++s) {
      const int k0 = (s & 3) * 32 + half * 8;
      A0[s] = mk_afrag((s < 4) ? rowi0 : rowj0, k0);
      A1[s] = mk_afrag((s < 4) ? rowi1 : rowj1, k0);
    }

    // basis preload: rows m and m+16, b-index = lane&3
    float bb0[8], bb1[8];
    #pragma unroll
    for (int j = 0; j < 8; ++j) {
      bb0[j] = basis[(base + j + half * 8) * 4 + (lane & 3)];
      bb1[j] = basis[(base + 16 + j + half * 8) * 4 + (lane & 3)];
    }

    // ---- GEMM1: H0[32x128] = X @ W0 + b0 (B fragment shared by both tiles) ----
    #pragma unroll
    for (int nt = 0; nt < 8; ++nt) {
      v8f c0, c1;
      #pragma unroll
      for (int q = 0; q < 8; ++q) { c0[q] = bias0[nt]; c1[q] = bias0[nt]; }
      const _Float16* bcol = &smem[W0_OFF + (16 * nt + r) * W0_STRIDE + half * 8];
      #pragma unroll
      for (int s = 0; s < 8; ++s) {
        v16h b = cat8(*(const v8h*)(bcol + 32 * s), *(const v8h*)(bcol + 32 * s + 16));
        c0 = __builtin_amdgcn_wmma_f32_16x16x32_f16(false, A0[s], false, b,
                                                    (short)0, c0, false, false);
        c1 = __builtin_amdgcn_wmma_f32_16x16x32_f16(false, A1[s], false, b,
                                                    (short)0, c1, false, false);
      }
      #pragma unroll
      for (int j = 0; j < 8; ++j) {
        hbuf[(j + half * 8) * H_STRIDE + 16 * nt + r]        = (_Float16)c0[j];
        hbuf[(16 + j + half * 8) * H_STRIDE + 16 * nt + r]   = (_Float16)c1[j];
      }
    }

    // ---- relayout H0 C->A via LDS ----
    v16h A20[4], A21[4];
    #pragma unroll
    for (int s = 0; s < 4; ++s) {
      const _Float16* a0 = &hbuf[r * H_STRIDE + 32 * s + half * 8];
      const _Float16* a1 = &hbuf[(16 + r) * H_STRIDE + 32 * s + half * 8];
      A20[s] = cat8(*(const v8h*)(a0), *(const v8h*)(a0 + 16));
      A21[s] = cat8(*(const v8h*)(a1), *(const v8h*)(a1 + 16));
    }

    // ---- GEMM2: H1[32x128] = H0 @ W1 + b1 (reuses hbuf) ----
    #pragma unroll
    for (int nt = 0; nt < 8; ++nt) {
      v8f c0, c1;
      #pragma unroll
      for (int q = 0; q < 8; ++q) { c0[q] = bias1[nt]; c1[q] = bias1[nt]; }
      const _Float16* bcol = &smem[W1_OFF + (16 * nt + r) * W1_STRIDE + half * 8];
      #pragma unroll
      for (int s = 0; s < 4; ++s) {
        v16h b = cat8(*(const v8h*)(bcol + 32 * s), *(const v8h*)(bcol + 32 * s + 16));
        c0 = __builtin_amdgcn_wmma_f32_16x16x32_f16(false, A20[s], false, b,
                                                    (short)0, c0, false, false);
        c1 = __builtin_amdgcn_wmma_f32_16x16x32_f16(false, A21[s], false, b,
                                                    (short)0, c1, false, false);
      }
      #pragma unroll
      for (int j = 0; j < 8; ++j) {
        hbuf[(j + half * 8) * H_STRIDE + 16 * nt + r]        = (_Float16)c0[j];
        hbuf[(16 + j + half * 8) * H_STRIDE + 16 * nt + r]   = (_Float16)c1[j];
      }
    }

    v16h A30[4], A31[4];
    #pragma unroll
    for (int s = 0; s < 4; ++s) {
      const _Float16* a0 = &hbuf[r * H_STRIDE + 32 * s + half * 8];
      const _Float16* a1 = &hbuf[(16 + r) * H_STRIDE + 32 * s + half * 8];
      A30[s] = cat8(*(const v8h*)(a0), *(const v8h*)(a0 + 16));
      A31[s] = cat8(*(const v8h*)(a1), *(const v8h*)(a1 + 16));
    }

    // ---- GEMM3 + basis contraction: out[e, c] = sum_b H2[e, 4c+b]*basis[e,b] ----
    #pragma unroll
    for (int nt = 0; nt < 16; ++nt) {
      v8f c0, c1;
      #pragma unroll
      for (int q = 0; q < 8; ++q) { c0[q] = 0.0f; c1[q] = 0.0f; }
      const _Float16* bcol = &smem[WO_OFF + (16 * nt + r) * WO_STRIDE + half * 8];
      #pragma unroll
      for (int s = 0; s < 4; ++s) {
        v16h b = cat8(*(const v8h*)(bcol + 32 * s), *(const v8h*)(bcol + 32 * s + 16));
        c0 = __builtin_amdgcn_wmma_f32_16x16x32_f16(false, A30[s], false, b,
                                                    (short)0, c0, false, false);
        c1 = __builtin_amdgcn_wmma_f32_16x16x32_f16(false, A31[s], false, b,
                                                    (short)0, c1, false, false);
      }
      const int c_col = 4 * nt + (r >> 2);
      #pragma unroll
      for (int j = 0; j < 8; ++j) {
        const int m = j + half * 8;
        float v0 = c0[j] * bb0[j];             // column n=16nt+r -> b = lane&3
        v0 += __shfl_xor(v0, 1, 32);
        v0 += __shfl_xor(v0, 2, 32);
        float v1 = c1[j] * bb1[j];
        v1 += __shfl_xor(v1, 1, 32);
        v1 += __shfl_xor(v1, 2, 32);
        if ((lane & 3) == 0) {
          out[(base + m) * OUT_CH + c_col]      = v0;
          out[(base + 16 + m) * OUT_CH + c_col] = v1;
        }
      }
    }
  }
}

extern "C" void kernel_launch(void* const* d_in, const int* in_sizes, int n_in,
                              void* d_out, int out_size, void* d_ws, size_t ws_size,
                              hipStream_t stream) {
  const float* prop  = (const float*)d_in[0];
  const int*   idx_i = (const int*)  d_in[1];
  const int*   idx_j = (const int*)  d_in[2];
  const float* basis = (const float*)d_in[3];
  const float* W0    = (const float*)d_in[4];
  const float* b0    = (const float*)d_in[5];
  const float* W1    = (const float*)d_in[6];
  const float* b1    = (const float*)d_in[7];
  const float* Wout  = (const float*)d_in[8];
  float* out = (float*)d_out;

  dim3 grid(512), block(256);
  hipLaunchKernelGGL(pilayer_kernel, grid, block, 0, stream,
                     prop, idx_i, idx_j, basis, W0, b0, W1, b1, Wout, out);
}
